// Model_67731634258522
// MI455X (gfx1250) — matrix operand order repaired and checked by reading, not compiled, not run on hardware
//
#include <hip/hip_runtime.h>
#include <hip/hip_bf16.h>
#include <math.h>

typedef __bf16 bf16_t;
typedef __attribute__((ext_vector_type(16))) __bf16 v16bf;
typedef __attribute__((ext_vector_type(8)))  __bf16 v8bf;
typedef __attribute__((ext_vector_type(8)))  float  v8f;

// ---------------------------------------------------------------------------
// Weight packing: OIHW fp32 -> [layer][tap][co_pad][ci] bf16 (ci contiguous,
// so a WMMA B fragment (32 ci x 16 co) is one 32B load per lane).
// ---------------------------------------------------------------------------
__global__ void pack_w_kernel(const float* __restrict__ src, bf16_t* __restrict__ dst,
                              int L, int Co, int CoPad, int Ci)
{
  long total = (long)L * 9 * CoPad * Ci;
  for (long i = (long)blockIdx.x * blockDim.x + threadIdx.x; i < total;
       i += (long)gridDim.x * blockDim.x) {
    int ci = (int)(i % Ci); long r = i / Ci;
    int o  = (int)(r % CoPad); r /= CoPad;
    int t  = (int)(r % 9); int l = (int)(r / 9);
    float v = 0.0f;
    if (o < Co) {
      int ky = t / 3, kx = t % 3;
      v = src[(((long)(l * Co + o) * Ci + ci) * 3 + ky) * 3 + kx];
    }
    dst[i] = (bf16_t)v;
  }
}

__global__ void pack_b_kernel(const float* __restrict__ src, float* __restrict__ dst,
                              int L, int Co, int CoPad)
{
  int total = L * CoPad;
  for (int i = blockIdx.x * blockDim.x + threadIdx.x; i < total;
       i += gridDim.x * blockDim.x) {
    int o = i % CoPad, l = i / CoPad;
    dst[i] = (o < Co) ? src[l * Co + o] : 0.0f;
  }
}

// NCHW fp32 -> NHWC bf16 (channels contiguous => WMMA A fragments contiguous)
__global__ void nchw_to_nhwc_kernel(const float* __restrict__ src, bf16_t* __restrict__ dst,
                                    int N, int C, int P /*H*W*/)
{
  long total = (long)N * P * C;
  for (long i = (long)blockIdx.x * blockDim.x + threadIdx.x; i < total;
       i += (long)gridDim.x * blockDim.x) {
    int c = (int)(i % C); long r = i / C;
    int p = (int)(r % P); int n = (int)(r / P);
    dst[i] = (bf16_t)src[((long)n * C + c) * P + p];
  }
}

// ---------------------------------------------------------------------------
// Implicit-GEMM 3x3 SAME conv using V_WMMA_F32_16X16X32_BF16.
//
// Tiling: one workgroup (128 thr = 4 waves) = one 32-pixel row segment
// (y0, x0..x0+31) x 256 output channels. Each wave: 32 pixels x 64 couts =
// 2 A fragments x 4 B fragments = 8 f32 accumulators, so every B fragment
// (the unique-per-wave operand) feeds two WMMAs and 8 independent WMMA
// chains hide the B load latency.
//
// The activation halo (3 rows x 34 cols x 32 ci) is staged in LDS by
// GLOBAL_LOAD_ASYNC_TO_LDS_B128, double-buffered across the 8 ci-blocks
// (ASYNCcnt pipelining). All 4 waves and all 9 taps read A fragments from
// LDS (ds_load_b128); B fragments stream from L2 via global_load_b128.
//
// LDS slot = one halo pixel's 32 ci (64B), padded to 80B to spread banks.
// mode: 0 = ReLU -> bf16 ; 1 = sigmoid -> bf16 ; 2 = raw -> f32
// ---------------------------------------------------------------------------
#define SLOT_B   80                     // 64B data + 16B bank pad
#define HALOC    34                     // halo columns: x0-1 .. x0+32
#define NSLOT    (3 * HALOC)            // 102 slots
#define BUF_B    (NSLOT * SLOT_B)       // 8160 B per ci-block buffer

__global__ __launch_bounds__(128)
void conv3x3_wmma_kernel(const bf16_t* __restrict__ act,   // [N][H][W][Cin]
                         const bf16_t* __restrict__ wpk,   // [9][CoutPad][Ci]
                         const float*  __restrict__ bias,  // [CoutPad]
                         bf16_t* __restrict__ out_bf,      // [N][H][W][CoutPad]
                         float*  __restrict__ out_f,
                         int H, int W, int Cin, int CoutPad, int mode)
{
  __shared__ __align__(16) char smem[2 * BUF_B];

  const int P      = H * W;
  const int lane   = threadIdx.x & 31;
  const int wave   = threadIdx.x >> 5;
  const int tilesX = (W + 31) >> 5;
  const int y0     = blockIdx.x / tilesX;
  const int x0     = (blockIdx.x % tilesX) << 5;
  const int n      = blockIdx.y;
  const int coBase = blockIdx.z * 256 + wave * 64;
  const bool coOK  = (coBase < CoutPad);          // uniform per wave

  const int m    = lane & 15;                     // pixel column / cout column
  const int half = lane >> 4;
  const size_t actN = (size_t)n * P * Cin;

  // --- async staging of the activation halo for one ci-block ---------------
  auto stage = [&](int ci0, int buf) {
    const unsigned base = (unsigned)(buf * BUF_B);
    for (int t = threadIdx.x; t < NSLOT * 4; t += 128) {   // 4 x 16B per slot
      int c    = t & 3;
      int slot = t >> 2;
      int dy   = slot / HALOC, dxi = slot % HALOC;
      int gy = y0 + dy - 1, gx = x0 + dxi - 1;
      unsigned loff = base + (unsigned)(slot * SLOT_B + c * 16);
      if (gy >= 0 && gy < H && gx >= 0 && gx < W) {
        unsigned long long ga = (unsigned long long)
            (act + actN + (size_t)(gy * W + gx) * Cin + ci0 + c * 8);
        asm volatile("global_load_async_to_lds_b128 %0, %1, off"
                     :: "v"(loff), "v"(ga) : "memory");
      } else {
        v8bf z = {};
        *(v8bf*)(smem + loff) = z;                 // zero-fill OOB halo
      }
    }
  };

  v8f acc0 = {}, acc1 = {}, acc2 = {}, acc3 = {};
  v8f acc4 = {}, acc5 = {}, acc6 = {}, acc7 = {};
  const bf16_t* wbase = wpk + (size_t)(coBase + m) * Cin + (size_t)half * 16;
  const int NCI = Cin >> 5;                        // 8 ci-blocks of 32

  stage(0, 0);
  asm volatile("s_wait_asynccnt 0x0" ::: "memory");
  __syncthreads();

  for (int k = 0; k < NCI; ++k) {
    const int ci0 = k << 5;
    const int buf = k & 1;
    if (k + 1 < NCI) stage(ci0 + 32, buf ^ 1);

    if (coOK) {
      const char* abase = smem + buf * BUF_B;
      __builtin_prefetch(wbase + ci0 + 32, 0, 0);  // gfx1250 global_prefetch_b8
#pragma unroll
      for (int tap = 0; tap < 9; ++tap) {
        const int ky = tap / 3, kx = tap % 3;      // 0..2 each
        // A0: pixels x0+m ; A1: pixels x0+16+m  (halo col = pixel + kx)
        const char* ap0 = abase + (ky * HALOC + (m + kx)) * SLOT_B + half * 32;
        const char* ap1 = ap0 + 16 * SLOT_B;
        v8bf a0lo = *(const v8bf*)(ap0);
        v8bf a0hi = *(const v8bf*)(ap0 + 16);
        v8bf a1lo = *(const v8bf*)(ap1);
        v8bf a1hi = *(const v8bf*)(ap1 + 16);
        v16bf A0 = __builtin_shufflevector(a0lo, a0hi,
                     0,1,2,3,4,5,6,7,8,9,10,11,12,13,14,15);
        v16bf A1 = __builtin_shufflevector(a1lo, a1hi,
                     0,1,2,3,4,5,6,7,8,9,10,11,12,13,14,15);
        const bf16_t* wp = wbase + (size_t)tap * CoutPad * Cin + ci0;
        v16bf B0 = *(const v16bf*)(wp);
        v16bf B1 = *(const v16bf*)(wp + (size_t)16 * Cin);
        v16bf B2 = *(const v16bf*)(wp + (size_t)32 * Cin);
        v16bf B3 = *(const v16bf*)(wp + (size_t)48 * Cin);
        acc0 = __builtin_amdgcn_wmma_f32_16x16x32_bf16(false, A0, false, B0, (short)0, acc0, false, false);
        acc4 = __builtin_amdgcn_wmma_f32_16x16x32_bf16(false, A1, false, B0, (short)0, acc4, false, false);
        acc1 = __builtin_amdgcn_wmma_f32_16x16x32_bf16(false, A0, false, B1, (short)0, acc1, false, false);
        acc5 = __builtin_amdgcn_wmma_f32_16x16x32_bf16(false, A1, false, B1, (short)0, acc5, false, false);
        acc2 = __builtin_amdgcn_wmma_f32_16x16x32_bf16(false, A0, false, B2, (short)0, acc2, false, false);
        acc6 = __builtin_amdgcn_wmma_f32_16x16x32_bf16(false, A1, false, B2, (short)0, acc6, false, false);
        acc3 = __builtin_amdgcn_wmma_f32_16x16x32_bf16(false, A0, false, B3, (short)0, acc3, false, false);
        acc7 = __builtin_amdgcn_wmma_f32_16x16x32_bf16(false, A1, false, B3, (short)0, acc7, false, false);
      }
    }

    asm volatile("s_wait_asynccnt 0x0" ::: "memory");
    __syncthreads();                               // staged data visible; LDS reads done
  }

  if (!coOK) return;

  float b0 = bias[coBase + m];
  float b1 = bias[coBase + 16 + m];
  float b2 = bias[coBase + 32 + m];
  float b3 = bias[coBase + 48 + m];

#pragma unroll
  for (int g = 0; g < 2; ++g) {                    // g=0: acc0..3, g=1: acc4..7
#pragma unroll
    for (int r = 0; r < 8; ++r) {
      int M = g * 16 + r + half * 8;               // C/D row layout
      int x = x0 + M;
      if (x >= W) continue;
      int pp = y0 * W + x;
      float v0, v1, v2, v3;
      if (g == 0) { v0 = acc0[r]; v1 = acc1[r]; v2 = acc2[r]; v3 = acc3[r]; }
      else        { v0 = acc4[r]; v1 = acc5[r]; v2 = acc6[r]; v3 = acc7[r]; }
      v0 += b0; v1 += b1; v2 += b2; v3 += b3;
      if (mode == 0) {
        v0 = fmaxf(v0, 0.f); v1 = fmaxf(v1, 0.f); v2 = fmaxf(v2, 0.f); v3 = fmaxf(v3, 0.f);
      } else if (mode == 1) {
        v0 = 1.0f / (1.0f + __expf(-v0)); v1 = 1.0f / (1.0f + __expf(-v1));
        v2 = 1.0f / (1.0f + __expf(-v2)); v3 = 1.0f / (1.0f + __expf(-v3));
      }
      if (mode == 2) {
        float* op = out_f + ((size_t)n * P + pp) * CoutPad + coBase + m;
        op[0] = v0; op[16] = v1; op[32] = v2; op[48] = v3;
      } else {
        bf16_t* op = out_bf + ((size_t)n * P + pp) * CoutPad + coBase + m;
        op[0] = (bf16_t)v0; op[16] = (bf16_t)v1; op[32] = (bf16_t)v2; op[48] = (bf16_t)v3;
      }
    }
  }
}

// ---------------------------------------------------------------------------
// Decode: histogram-select top-k (approx), then candidate collection with
// fused delta2box.
// ---------------------------------------------------------------------------
__global__ void hist_kernel(const bf16_t* __restrict__ cls, int P, int CoPad, int Creal,
                            unsigned* __restrict__ hist /*[8][1024]*/)
{
  long total = 8L * P * Creal;
  for (long i = (long)blockIdx.x * blockDim.x + threadIdx.x; i < total;
       i += (long)gridDim.x * blockDim.x) {
    int c = (int)(i % Creal); long r = i / Creal;
    int p = (int)(r % P); int n = (int)(r / P);
    float s = (float)cls[((size_t)n * P + p) * CoPad + c];
    if (s >= 0.05f) {
      int b = (int)(s * 1024.0f); if (b > 1023) b = 1023;
      atomicAdd(&hist[n * 1024 + b], 1u);
    }
  }
}

__global__ void cutoff_kernel(const unsigned* __restrict__ hist, float* __restrict__ cut)
{
  if (threadIdx.x != 0) return;
  int n = blockIdx.x;
  unsigned acc = 0;
  float c = 0.05f;
  for (int b = 1023; b >= 0; --b) {
    acc += hist[n * 1024 + b];
    if (acc >= 1000u) { c = fmaxf(0.05f, (float)b / 1024.0f); break; }
  }
  cut[n] = c;
}

__device__ __forceinline__ void anchor4(int a, float stride, float* an)
{
  int ri = a % 3;
  float ratio = (ri == 0) ? 1.0f : (ri == 1 ? 2.0f : 0.5f);
  float scale = 4.0f * exp2f((float)(a / 3) * (1.0f / 3.0f));
  float wsr   = sqrtf(stride * stride / ratio);
  float dw = wsr, dh = wsr * ratio;
  an[0] = 0.5f * (stride - dw * scale);
  an[1] = 0.5f * (stride - dh * scale);
  an[2] = 0.5f * (stride + dw * scale) - 1.0f;
  an[3] = 0.5f * (stride + dh * scale) - 1.0f;
}

__global__ void collect_kernel(const bf16_t* __restrict__ cls, const float* __restrict__ boxd,
                               const float* __restrict__ cut, int* __restrict__ cnt,
                               float* __restrict__ cs, float* __restrict__ cb, float* __restrict__ cc,
                               int H, int W, int lvl, const int* __restrict__ image_size)
{
  int P = H * W;
  float stride = (float)(*image_size) / (float)W;
  long total = 8L * P * 720;
  for (long i = (long)blockIdx.x * blockDim.x + threadIdx.x; i < total;
       i += (long)gridDim.x * blockDim.x) {
    int ch = (int)(i % 720); long r = i / 720;
    int p  = (int)(r % P);   int n = (int)(r / P);
    float s = (float)cls[((size_t)n * P + p) * 768 + ch];
    if (s < 0.05f || s < cut[n]) continue;
    int pos = atomicAdd(&cnt[n], 1);
    if (pos >= 1000) continue;
    int a = ch / 80, c = ch % 80;
    int x = p % W, y = p / W;
    const float* bd = boxd + ((size_t)n * P + p) * 64 + a * 4;
    float d0 = bd[0], d1 = bd[1], d2 = bd[2], d3 = bd[3];
    float an[4]; anchor4(a, stride, an);
    float gx = x * stride, gy = y * stride;
    float ax1 = gx + an[0], ay1 = gy + an[1], ax2 = gx + an[2], ay2 = gy + an[3];
    float w = ax2 - ax1 + 1.0f, h = ay2 - ay1 + 1.0f;
    float cx = ax1 + 0.5f * w, cy = ay1 + 0.5f * h;
    float pcx = d0 * w + cx, pcy = d1 * h + cy;
    float pw = __expf(d2) * w, ph = __expf(d3) * h;
    float Mx = (float)W * stride - 1.0f, My = (float)H * stride - 1.0f;
    float bx1 = fminf(fmaxf(pcx - 0.5f * pw, 0.f), Mx);
    float by1 = fminf(fmaxf(pcy - 0.5f * ph, 0.f), My);
    float bx2 = fminf(fmaxf(pcx + 0.5f * pw - 1.f, 0.f), Mx);
    float by2 = fminf(fmaxf(pcy + 0.5f * ph - 1.f, 0.f), My);
    size_t slot = ((size_t)n * 5 + lvl) * 1000 + pos;
    cs[slot] = s; cc[slot] = (float)c;
    cb[slot * 4 + 0] = bx1; cb[slot * 4 + 1] = by1;
    cb[slot * 4 + 2] = bx2; cb[slot * 4 + 3] = by2;
  }
}

// ---------------------------------------------------------------------------
// Greedy NMS: one block per image over 5000 candidates, 100 detections.
// d_out = scores[8][100] ++ boxes[8][100][4] ++ classes[8][100]
// ---------------------------------------------------------------------------
__global__ __launch_bounds__(256)
void nms_kernel(float* __restrict__ cs, const float* __restrict__ cb,
                const float* __restrict__ cc, float* __restrict__ out)
{
  const int n   = blockIdx.x;
  const int tid = threadIdx.x;
  float* s = cs + (size_t)n * 5000;
  const float* b = cb + (size_t)n * 5000 * 4;
  const float* k = cc + (size_t)n * 5000;
  __shared__ float sv[256];
  __shared__ int   si[256];
  __shared__ float bsm; __shared__ int bim;

  for (int det = 0; det < 100; ++det) {
    float best = -1.0f; int bi = -1;
    for (int j = tid; j < 5000; j += 256) {
      float v = s[j];
      if (v > best) { best = v; bi = j; }
    }
    sv[tid] = best; si[tid] = bi;
    __syncthreads();
    for (int o = 128; o > 0; o >>= 1) {
      if (tid < o && sv[tid + o] > sv[tid]) { sv[tid] = sv[tid + o]; si[tid] = si[tid + o]; }
      __syncthreads();
    }
    if (tid == 0) {
      bsm = sv[0]; bim = si[0];
      int im = bim;
      if (bsm > 0.0f && im >= 0) {
        out[n * 100 + det] = bsm;
        out[800 + (n * 100 + det) * 4 + 0] = b[im * 4 + 0];
        out[800 + (n * 100 + det) * 4 + 1] = b[im * 4 + 1];
        out[800 + (n * 100 + det) * 4 + 2] = b[im * 4 + 2];
        out[800 + (n * 100 + det) * 4 + 3] = b[im * 4 + 3];
        out[4000 + n * 100 + det] = k[im];
        s[im] = 0.0f;
      } else {
        out[n * 100 + det] = 0.0f;
        out[800 + (n * 100 + det) * 4 + 0] = 0.0f;
        out[800 + (n * 100 + det) * 4 + 1] = 0.0f;
        out[800 + (n * 100 + det) * 4 + 2] = 0.0f;
        out[800 + (n * 100 + det) * 4 + 3] = 0.0f;
        out[4000 + n * 100 + det] = 0.0f;
      }
    }
    __syncthreads();
    float sm = bsm; int im = bim;
    if (sm > 0.0f && im >= 0) {
      float bx1 = b[im * 4], by1 = b[im * 4 + 1], bx2 = b[im * 4 + 2], by2 = b[im * 4 + 3];
      float ai  = (bx2 - bx1 + 1.0f) * (by2 - by1 + 1.0f);
      float ki  = k[im];
      for (int j = tid; j < 5000; j += 256) {
        float sj = s[j];
        if (sj <= 0.0f || k[j] != ki) continue;
        float xx1 = fmaxf(b[j * 4], bx1),     yy1 = fmaxf(b[j * 4 + 1], by1);
        float xx2 = fminf(b[j * 4 + 2], bx2), yy2 = fminf(b[j * 4 + 3], by2);
        float iw = fmaxf(xx2 - xx1 + 1.0f, 0.0f), ih = fmaxf(yy2 - yy1 + 1.0f, 0.0f);
        float inter = iw * ih;
        float aj = (b[j * 4 + 2] - b[j * 4] + 1.0f) * (b[j * 4 + 3] - b[j * 4 + 1] + 1.0f);
        float iou = inter / (ai + aj - inter);
        if (iou > 0.5f) s[j] = 0.0f;
      }
    }
    __syncthreads();
  }
}

// ---------------------------------------------------------------------------
// Host orchestration
// ---------------------------------------------------------------------------
extern "C" void kernel_launch(void* const* d_in, const int* in_sizes, int n_in,
                              void* d_out, int out_size, void* d_ws, size_t ws_size,
                              hipStream_t stream)
{
  (void)in_sizes; (void)n_in; (void)out_size; (void)ws_size;
  static const int SIZES[5] = {96, 48, 24, 12, 6};
  const float* feat[5];
  for (int i = 0; i < 5; ++i) feat[i] = (const float*)d_in[i];
  const float* cls_wh = (const float*)d_in[5];
  const float* cls_bh = (const float*)d_in[6];
  const float* cls_wo = (const float*)d_in[7];
  const float* cls_bo = (const float*)d_in[8];
  const float* box_wh = (const float*)d_in[9];
  const float* box_bh = (const float*)d_in[10];
  const float* box_wo = (const float*)d_in[11];
  const float* box_bo = (const float*)d_in[12];
  const int* image_size = (const int*)d_in[13];

  char* ws = (char*)d_ws;
  size_t off = 0;
  auto alloc = [&](size_t bytes) -> size_t {
    size_t o = off; off += (bytes + 255) & ~(size_t)255; return o;
  };
  const size_t PMAX = 96 * 96;
  size_t o_actA = alloc(8ull * PMAX * 256 * 2);
  size_t o_actB = alloc(8ull * PMAX * 256 * 2);
  size_t o_actC = alloc(8ull * PMAX * 256 * 2);
  size_t o_wch  = alloc(4ull * 9 * 256 * 256 * 2);
  size_t o_wbh  = alloc(4ull * 9 * 256 * 256 * 2);
  size_t o_wco  = alloc(9ull * 768 * 256 * 2);
  size_t o_wbo  = alloc(9ull * 64 * 256 * 2);
  size_t o_bch  = alloc(4 * 256 * 4);
  size_t o_bbh  = alloc(4 * 256 * 4);
  size_t o_bco  = alloc(768 * 4);
  size_t o_bbo  = alloc(64 * 4);
  size_t o_clss = alloc(8ull * PMAX * 768 * 2);  // bf16 sigmoid scores
  size_t o_boxd = alloc(8ull * PMAX * 64 * 4);   // f32 box deltas
  size_t o_hist = alloc(8 * 1024 * 4);
  size_t o_cut  = alloc(8 * 4);
  size_t o_cnt  = alloc(8 * 4);
  size_t o_cs   = alloc(8ull * 5 * 1000 * 4);
  size_t o_cb   = alloc(8ull * 5 * 1000 * 4 * 4);
  size_t o_cc   = alloc(8ull * 5 * 1000 * 4);

  bf16_t* actA = (bf16_t*)(ws + o_actA);
  bf16_t* actB = (bf16_t*)(ws + o_actB);
  bf16_t* actC = (bf16_t*)(ws + o_actC);
  bf16_t* wch  = (bf16_t*)(ws + o_wch);
  bf16_t* wbh  = (bf16_t*)(ws + o_wbh);
  bf16_t* wco  = (bf16_t*)(ws + o_wco);
  bf16_t* wbo  = (bf16_t*)(ws + o_wbo);
  float*  bch  = (float*)(ws + o_bch);
  float*  bbh  = (float*)(ws + o_bbh);
  float*  bco  = (float*)(ws + o_bco);
  float*  bbo  = (float*)(ws + o_bbo);
  bf16_t* clss = (bf16_t*)(ws + o_clss);
  float*  boxd = (float*)(ws + o_boxd);
  unsigned* hist = (unsigned*)(ws + o_hist);
  float*  cutf = (float*)(ws + o_cut);
  int*    cnt  = (int*)(ws + o_cnt);
  float*  cs   = (float*)(ws + o_cs);
  float*  cb   = (float*)(ws + o_cb);
  float*  cc   = (float*)(ws + o_cc);

  // Zero candidate buffers and output (rebuilt fully every call)
  hipMemsetAsync(cs, 0, 8ull * 5 * 1000 * 4, stream);
  hipMemsetAsync(cb, 0, 8ull * 5 * 1000 * 16, stream);
  hipMemsetAsync(cc, 0, 8ull * 5 * 1000 * 4, stream);
  hipMemsetAsync(d_out, 0, 4800 * sizeof(float), stream);

  // Pack weights / biases to WMMA-friendly bf16 layouts
  pack_w_kernel<<<2048, 256, 0, stream>>>(cls_wh, wch, 4, 256, 256, 256);
  pack_w_kernel<<<2048, 256, 0, stream>>>(box_wh, wbh, 4, 256, 256, 256);
  pack_w_kernel<<<1024, 256, 0, stream>>>(cls_wo, wco, 1, 720, 768, 256);
  pack_w_kernel<<<128, 256, 0, stream>>>(box_wo, wbo, 1, 36, 64, 256);
  pack_b_kernel<<<8, 256, 0, stream>>>(cls_bh, bch, 4, 256, 256);
  pack_b_kernel<<<8, 256, 0, stream>>>(box_bh, bbh, 4, 256, 256);
  pack_b_kernel<<<4, 256, 0, stream>>>(cls_bo, bco, 1, 720, 768);
  pack_b_kernel<<<1, 64, 0, stream>>>(box_bo, bbo, 1, 36, 64);

  const size_t WLAYER = 9ull * 256 * 256;  // elements per hidden layer weight pack

  for (int l = 0; l < 5; ++l) {
    int H = SIZES[l], W = SIZES[l], P = H * W;
    int tilesX = (W + 31) / 32;
    nchw_to_nhwc_kernel<<<1024, 256, 0, stream>>>(feat[l], actA, 8, 256, P);

    dim3 cg(tilesX * H, 8, 1);   // Cout 256
    dim3 og(tilesX * H, 8, 3);   // Cout 768

    // --- cls head: 4 hidden convs (ReLU) + output conv (sigmoid) ---
    conv3x3_wmma_kernel<<<cg, 128, 0, stream>>>(actA, wch + 0 * WLAYER, bch + 0 * 256, actB, nullptr, H, W, 256, 256, 0);
    conv3x3_wmma_kernel<<<cg, 128, 0, stream>>>(actB, wch + 1 * WLAYER, bch + 1 * 256, actC, nullptr, H, W, 256, 256, 0);
    conv3x3_wmma_kernel<<<cg, 128, 0, stream>>>(actC, wch + 2 * WLAYER, bch + 2 * 256, actB, nullptr, H, W, 256, 256, 0);
    conv3x3_wmma_kernel<<<cg, 128, 0, stream>>>(actB, wch + 3 * WLAYER, bch + 3 * 256, actC, nullptr, H, W, 256, 256, 0);
    conv3x3_wmma_kernel<<<og, 128, 0, stream>>>(actC, wco, bco, clss, nullptr, H, W, 256, 768, 1);

    // --- box head: 4 hidden convs (ReLU) + output conv (raw f32) ---
    conv3x3_wmma_kernel<<<cg, 128, 0, stream>>>(actA, wbh + 0 * WLAYER, bbh + 0 * 256, actB, nullptr, H, W, 256, 256, 0);
    conv3x3_wmma_kernel<<<cg, 128, 0, stream>>>(actB, wbh + 1 * WLAYER, bbh + 1 * 256, actC, nullptr, H, W, 256, 256, 0);
    conv3x3_wmma_kernel<<<cg, 128, 0, stream>>>(actC, wbh + 2 * WLAYER, bbh + 2 * 256, actB, nullptr, H, W, 256, 256, 0);
    conv3x3_wmma_kernel<<<cg, 128, 0, stream>>>(actB, wbh + 3 * WLAYER, bbh + 3 * 256, actC, nullptr, H, W, 256, 256, 0);
    conv3x3_wmma_kernel<<<cg, 128, 0, stream>>>(actC, wbo, bbo, nullptr, boxd, H, W, 256, 64, 2);

    // --- decode this level ---
    hipMemsetAsync(hist, 0, 8 * 1024 * 4, stream);
    hipMemsetAsync(cnt, 0, 8 * 4, stream);
    hist_kernel<<<1024, 256, 0, stream>>>(clss, P, 768, 720, hist);
    cutoff_kernel<<<8, 32, 0, stream>>>(hist, cutf);
    collect_kernel<<<1024, 256, 0, stream>>>(clss, boxd, cutf, cnt, cs, cb, cc,
                                             H, W, l, image_size);
  }

  nms_kernel<<<8, 256, 0, stream>>>(cs, cb, cc, (float*)d_out);
}
